// FrequencyAwareNTXentLoss_91164975825467
// MI455X (gfx1250) — compile-verified, matrix operand order checked by reading, not run on hardware
//
#include <hip/hip_runtime.h>
#include <math.h>

#define BB 1024
#define NN 2048
#define DREP 128
#define TLEN 3840
#define NANCH 9
#define NSEG 29
#define NBIN 90            // freq bins k = 1..90 (0.5..45 Hz)
#define NTASK (NBIN*NSEG)  // 2610
#define INVT 5.0f          // 1/TEMP
#define EPSV 1e-8f
#define PSD_SCALE (1.0f/178176.0f) // 2 * (1/(128*96)) / 29
#define TWO_PI_256 0.0245436926061702596f // 2*pi/256

typedef float  v2f  __attribute__((ext_vector_type(2)));
typedef float  v8f  __attribute__((ext_vector_type(8)));
typedef _Float16 v16h __attribute__((ext_vector_type(16)));

// ---------------- K1: row-normalize reps = [zjs; zis] ----------------
__global__ void norm_k(const float* __restrict__ zis, const float* __restrict__ zjs,
                       float* __restrict__ rn) {
  int r = blockIdx.x, t = threadIdx.x;
  const float* src = (r < BB) ? (zjs + r * DREP) : (zis + (r - BB) * DREP);
  __shared__ float red[128];
  float v = src[t];
  red[t] = v * v;
  __syncthreads();
  for (int off = 64; off > 0; off >>= 1) {
    if (t < off) red[t] += red[t + off];
    __syncthreads();
  }
  float nrm = fmaxf(sqrtf(red[0]), EPSV);
  rn[r * DREP + t] = v / nrm;
}

// ---------------- K2: sim = rn @ rn^T via WMMA (one wave per 16x16 tile) ----
__global__ void sim_k(const float* __restrict__ rn, float* __restrict__ sim) {
  int tn = blockIdx.x, tm = blockIdx.y;
  int l = threadIdx.x;      // 0..31, one full wave, EXEC all ones
  int half = l >> 4;        // 0 or 1
  int lrow = l & 15;
  const float* arow = rn + (size_t)(tm * 16 + lrow) * DREP;
  const float* brow = rn + (size_t)(tn * 16 + lrow) * DREP;
  v8f c = {};
#if __has_builtin(__builtin_amdgcn_wmma_f32_16x16x4_f32)
  // 32-bit A 16x4: lane holds row M=l&15; VGPR v + lane-half select K = 2*half + v.
  // B(4x16) mirrors A with N = l&15. Chain 32 WMMAs over K=128, f32 exact.
  #pragma unroll
  for (int kc = 0; kc < DREP; kc += 4) {
    int kb = kc + (half << 1);
    v2f a = *(const v2f*)(arow + kb);
    v2f b = *(const v2f*)(brow + kb);
    c = __builtin_amdgcn_wmma_f32_16x16x4_f32(false, a, false, b, (short)0, c,
                                              false, false);
  }
#else
  // Fallback: f16 16x16x32 (codegen-confirmed). 16-bit A 16x32 layout:
  // lane-half selects K base = 8*half; elems 0..7 -> K=base+0..7, 8..15 -> K=base+16..23.
  #pragma unroll
  for (int kc = 0; kc < DREP; kc += 32) {
    v16h a, b;
    int base = kc + half * 8;
    #pragma unroll
    for (int e = 0; e < 8; ++e) {
      a[e]     = (_Float16)arow[base + e];
      a[e + 8] = (_Float16)arow[base + 16 + e];
      b[e]     = (_Float16)brow[base + e];
      b[e + 8] = (_Float16)brow[base + 16 + e];
    }
    c = __builtin_amdgcn_wmma_f32_16x16x32_f16(false, a, false, b, (short)0, c,
                                               false, false);
  }
#endif
  int col = tn * 16 + lrow;
  float* out = sim + (size_t)(tm * 16 + half * 8) * NN + col;
  #pragma unroll
  for (int v = 0; v < 8; ++v) out[(size_t)v * NN] = c[v];
}

// ---------------- K3: fused anchors-mean + Welch PSD + band powers ----------
// One block per signal. Streams anchors exactly once (141 MB total => HBM bound).
__global__ void welch_k(const float* __restrict__ anchors,
                        float* __restrict__ bp, float* __restrict__ sq) {
  int s = blockIdx.x, tid = threadIdx.x; // 256 threads
  __shared__ float xw[NSEG * 256];   // demeaned*windowed segments (7424)
  __shared__ float xbuf[TLEN];       // averaged signal (3840)
  __shared__ float ctab[256], stab[256];
  __shared__ float means[NSEG];
  __shared__ float partial[NTASK];   // per (bin,seg) |spec|^2 (2610)
  __shared__ float psd[NBIN + 2];

  // 9-anchor mean, fused (no avg materialization)
  const float* base = anchors + (size_t)s * NANCH * TLEN;
  for (int i = tid; i < TLEN; i += 256) {
    float acc = 0.f;
    #pragma unroll
    for (int a = 0; a < NANCH; ++a) acc += base[(size_t)a * TLEN + i];
    xbuf[i] = acc * (1.0f / 9.0f);
  }
  // twiddle table: ctab[m]=cos(2*pi*m/256); also provides the Hann window.
  {
    float ang = TWO_PI_256 * (float)tid;
    ctab[tid] = cosf(ang);
    stab[tid] = sinf(ang);
  }
  __syncthreads();
  if (tid < NSEG) {
    float acc = 0.f;
    const float* xp = xbuf + tid * 128;
    for (int t = 0; t < 256; ++t) acc += xp[t];
    means[tid] = acc * (1.0f / 256.0f);
  }
  __syncthreads();
  for (int idx = tid; idx < NSEG * 256; idx += 256) {
    int seg = idx >> 8, t = idx & 255;
    float w = 0.5f - 0.5f * ctab[t]; // Hann
    xw[idx] = (xbuf[seg * 128 + t] - means[seg]) * w;
  }
  __syncthreads();
  // 90-bin DFT per segment: spec_k = sum_t xw[t] * e^{-i*2*pi*k*t/256}
  for (int task = tid; task < NTASK; task += 256) {
    int kx = task / NSEG;
    int seg = task - kx * NSEG;
    int k = kx + 1;
    const float* xp = xw + seg * 256;
    float re = 0.f, im = 0.f;
    int m = 0;
    for (int t = 0; t < 256; ++t) {
      float v = xp[t];
      re += v * ctab[m];
      im += v * stab[m];
      m = (m + k) & 255;
    }
    partial[task] = re * re + im * im;
  }
  __syncthreads();
  if (tid < NBIN) {
    float p = 0.f;
    for (int seg = 0; seg < NSEG; ++seg) p += partial[tid * NSEG + seg];
    psd[tid + 1] = p * PSD_SCALE;
  }
  __syncthreads();
  if (tid == 0) {
    const int klo[5] = {1, 8, 16, 26, 60};
    const int khi[5] = {8, 16, 26, 60, 90};
    float bpl[5], tot = 0.f;
    for (int b = 0; b < 5; ++b) {
      float v = 0.f;
      for (int k = klo[b]; k < khi[b]; ++k) v += 0.25f * (psd[k] + psd[k + 1]);
      bpl[b] = v;
      tot += v;
    }
    float acc = 0.f;
    for (int b = 0; b < 5; ++b) {
      float bv = bpl[b] / tot;
      bp[s * 5 + b] = bv;
      acc += bv * bv;
    }
    sq[s] = acc;
  }
}

// ---------------- K4: pairwise distance matrix (1024x1024, 5-dim) -----------
__global__ void dist_k(const float* __restrict__ bp, const float* __restrict__ sq,
                       float* __restrict__ dist) {
  int idx = blockIdx.x * 256 + threadIdx.x;
  int i = idx >> 10, j = idx & 1023;
  const float* bi = bp + i * 5;
  const float* bj = bp + j * 5;
  float dot = 0.f;
  #pragma unroll
  for (int b = 0; b < 5; ++b) dot += bi[b] * bj[b];
  float d2 = sq[i] + sq[j] - 2.0f * dot;
  dist[idx] = sqrtf(fmaxf(d2, 0.0f));
}

// ---------------- K5: per-row bitonic sort -> quartiles ---------------------
__global__ void quant_k(const float* __restrict__ dist,
                        float* __restrict__ loq, float* __restrict__ hiq) {
  int r = blockIdx.x, tid = threadIdx.x; // 256
  __shared__ float s[1024];
  for (int i = tid; i < 1024; i += 256) s[i] = dist[(size_t)r * 1024 + i];
  __syncthreads();
  for (int k = 2; k <= 1024; k <<= 1) {
    for (int j = k >> 1; j > 0; j >>= 1) {
      for (int i = tid; i < 1024; i += 256) {
        int ixj = i ^ j;
        if (ixj > i) {
          float a = s[i], b = s[ixj];
          bool asc = ((i & k) == 0);
          if ((a > b) == asc) { s[i] = b; s[ixj] = a; }
        }
      }
      __syncthreads();
    }
  }
  if (tid == 0) {
    loq[r] = 0.25f * s[255] + 0.75f * s[256]; // quantile(0.25), linear interp
    hiq[r] = 0.75f * s[767] + 0.25f * s[768]; // quantile(0.75)
  }
}

// ---------------- K6: top-5 of IQR-masked dist (lax.top_k tie-break) --------
__global__ void topk_k(const float* __restrict__ dist, const float* __restrict__ loq,
                       const float* __restrict__ hiq, int* __restrict__ hard) {
  int r = blockIdx.x, tid = threadIdx.x; // 256
  float lo = loq[r], hi = hiq[r];
  float vals[4];
  #pragma unroll
  for (int q = 0; q < 4; ++q) {
    int j = tid + q * 256;
    float d = dist[(size_t)r * 1024 + j];
    vals[q] = (d >= lo && d <= hi) ? d : -__builtin_inff();
  }
  __shared__ float bv[256];
  __shared__ int bi[256];
  for (int round = 0; round < 5; ++round) {
    float bestv = vals[0];
    int besti = tid;
    #pragma unroll
    for (int q = 1; q < 4; ++q) { // ascending index order: strict > keeps min index
      if (vals[q] > bestv) { bestv = vals[q]; besti = tid + q * 256; }
    }
    bv[tid] = bestv; bi[tid] = besti;
    __syncthreads();
    for (int off = 128; off > 0; off >>= 1) {
      if (tid < off) {
        float ov = bv[tid + off]; int oi = bi[tid + off];
        if (ov > bv[tid] || (ov == bv[tid] && oi < bi[tid])) { bv[tid] = ov; bi[tid] = oi; }
      }
      __syncthreads();
    }
    int win = bi[0];
    if (tid == 0) hard[r * 5 + round] = win;
    if ((win & 255) == tid) vals[win >> 8] = -__builtin_inff();
    __syncthreads();
  }
}

// ---------------- K7: per-row logsumexp loss --------------------------------
// NEG_COLS mapping is analytic: with excluded cols e0<e1, negatives-index j
// maps to sim column j + (j>=e0) + (j>=e1-1).
__global__ void rowloss_k(const float* __restrict__ sim, const int* __restrict__ hard,
                          float* __restrict__ rowloss) {
  int r = blockIdx.x, tid = threadIdx.x; // 256
  int pcol = (r + BB) & (NN - 1);
  int e0 = min(r, pcol), e1 = max(r, pcol);
  __shared__ float srow[NN];
  __shared__ int hc[5];
  __shared__ float red[256];
  for (int c = tid; c < NN; c += 256) srow[c] = sim[(size_t)r * NN + c];
  if (tid < 5) {
    int h = hard[(r & (BB - 1)) * 5 + tid];
    hc[tid] = h + (h >= e0 ? 1 : 0) + (h >= e1 - 1 ? 1 : 0);
  }
  __syncthreads();
  int h0 = hc[0], h1 = hc[1], h2 = hc[2], h3 = hc[3], h4 = hc[4];
  float pos = srow[pcol] * INVT;
  float m = pos;
  for (int c = tid; c < NN; c += 256) {
    if (c == r || c == pcol) continue;
    float f = (c == h0 || c == h1 || c == h2 || c == h3 || c == h4) ? 3.0f : 1.0f;
    m = fmaxf(m, srow[c] * f * INVT);
  }
  red[tid] = m;
  __syncthreads();
  for (int off = 128; off > 0; off >>= 1) {
    if (tid < off) red[tid] = fmaxf(red[tid], red[tid + off]);
    __syncthreads();
  }
  float M = red[0];
  __syncthreads();
  float ssum = (tid == 0) ? expf(pos - M) : 0.f;
  for (int c = tid; c < NN; c += 256) {
    if (c == r || c == pcol) continue;
    float f = (c == h0 || c == h1 || c == h2 || c == h3 || c == h4) ? 3.0f : 1.0f;
    ssum += expf(srow[c] * f * INVT - M);
  }
  red[tid] = ssum;
  __syncthreads();
  for (int off = 128; off > 0; off >>= 1) {
    if (tid < off) red[tid] += red[tid + off];
    __syncthreads();
  }
  if (tid == 0) rowloss[r] = M + logf(red[0]) - pos;
}

// ---------------- K8: final deterministic reduction -------------------------
__global__ void final_k(const float* __restrict__ rowloss, float* __restrict__ out) {
  int tid = threadIdx.x;
  __shared__ float red[256];
  float acc = 0.f;
  for (int i = tid; i < NN; i += 256) acc += rowloss[i];
  red[tid] = acc;
  __syncthreads();
  for (int off = 128; off > 0; off >>= 1) {
    if (tid < off) red[tid] += red[tid + off];
    __syncthreads();
  }
  if (tid == 0) out[0] = red[0] / (float)NN;
}

extern "C" void kernel_launch(void* const* d_in, const int* in_sizes, int n_in,
                              void* d_out, int out_size, void* d_ws, size_t ws_size,
                              hipStream_t stream) {
  const float* zis = (const float*)d_in[0];
  const float* zjs = (const float*)d_in[1];
  const float* anchors = (const float*)d_in[2];
  float* W = (float*)d_ws;
  float* rn      = W;                 // 2048*128        = 262144
  float* sim     = rn + 262144;       // 2048*2048       = 4194304
  float* bp      = sim + 4194304;     // 1024*5
  float* sq      = bp + 5120;         // 1024
  float* dist    = sq + 1024;         // 1024*1024       = 1048576
  float* loq     = dist + 1048576;    // 1024
  float* hiq     = loq + 1024;        // 1024
  int*   hard    = (int*)(hiq + 1024);// 1024*5 ints
  float* rowloss = (float*)(hard + 5120); // 2048

  norm_k   <<<NN, 128, 0, stream>>>(zis, zjs, rn);
  sim_k    <<<dim3(NN / 16, NN / 16), 32, 0, stream>>>(rn, sim);
  welch_k  <<<BB, 256, 0, stream>>>(anchors, bp, sq);
  dist_k   <<<(BB * BB) / 256, 256, 0, stream>>>(bp, sq, dist);
  quant_k  <<<BB, 256, 0, stream>>>(dist, loq, hiq);
  topk_k   <<<BB, 256, 0, stream>>>(dist, loq, hiq, hard);
  rowloss_k<<<NN, 256, 0, stream>>>(sim, hard, rowloss);
  final_k  <<<1, 256, 0, stream>>>(rowloss, (float*)d_out);
}
